// BertSelfAttentionWithRelation_40209483825289
// MI455X (gfx1250) — matrix operand order repaired and checked by reading, not compiled, not run on hardware
//
#include <hip/hip_runtime.h>
#include <math.h>

// ---------------------------------------------------------------------------
// BertSelfAttentionWithRelation for MI455X (gfx1250, wave32, WMMA + TDM)
// B=4, L=512, H=12, D=64, HIDDEN=768. fp32 in/out, f16 WMMA w/ f32 accumulate.
// ---------------------------------------------------------------------------

typedef __attribute__((ext_vector_type(16))) _Float16 v16h;
typedef __attribute__((ext_vector_type(8)))  float    v8f;
typedef __attribute__((ext_vector_type(4)))  unsigned int v4u;
typedef __attribute__((ext_vector_type(8)))  int      v8i;
typedef __attribute__((ext_vector_type(4)))  int      v4i;

constexpr int BATCH = 4;
constexpr int L     = 512;
constexpr int H     = 12;
constexpr int D     = 64;
constexpr int HID   = 768;  // H*D

// ---- WMMA fragment loaders (f32 memory -> f16 fragments, CDNA5 wave32) ----

// A fragment: 16x32 (MxK) from row-major source, row stride rstride (elems),
// K offset koff. Rows >= mvalid are zero-filled (head padding 12 -> 16).
// Layout: lanes 0-15 M=lane (K 0..15 in elems 0..15);
// lanes 16-31 M=lane-16 (K 16..31), matching the B fragments.
__device__ inline v16h load_a(const float* p, long rstride, int koff,
                              int lane, int mvalid) {
  const int m    = lane & 15;
  const int half = lane >> 4;
  v16h a;
  if (m < mvalid) {
    const float* row = p + (size_t)m * rstride + koff;
#pragma unroll
    for (int e = 0; e < 16; ++e) a[e] = (_Float16)row[half * 16 + e];
  } else {
#pragma unroll
    for (int e = 0; e < 16; ++e) a[e] = (_Float16)0.0f;
  }
  return a;
}

// B fragment: 32x16 (KxN) from row-major [K x N] source, leading dim ld.
// Layout: lane holds column (lane&15); K = (lane>>4)*16 + elem.
__device__ inline v16h load_b(const float* p, long ld, int lane) {
  const int n  = lane & 15;
  const int kb = (lane >> 4) * 16;
  v16h b;
#pragma unroll
  for (int e = 0; e < 16; ++e) b[e] = (_Float16)p[(size_t)(kb + e) * ld + n];
  return b;
}

// B fragment where B(k,n) = src[n*ld + k]  (transposed read of row-major [N x K]).
__device__ inline v16h load_bt(const float* p, long ld, int lane) {
  const int n  = lane & 15;
  const int kb = (lane >> 4) * 16;
  v16h b;
#pragma unroll
  for (int e = 0; e < 16; ++e) b[e] = (_Float16)p[(size_t)n * ld + kb + e];
  return b;
}

__device__ inline v8f wmma_step(v16h a, v16h b, v8f c) {
  // D = A(16x32,f16) * B(32x16,f16) + C(16x16,f32)
  return __builtin_amdgcn_wmma_f32_16x16x32_f16(false, a, false, b,
                                                (short)0, c, false, false);
}

// ---------------------------------------------------------------------------
// K1: QKV projection.  dst[b,h,l,d] = (x @ W + bias) reshaped to heads.
// x: [B*L, HID] row-major; W: [HID, HID] row-major (in x out).
// grid (HID/64, B*L/16), block (32,4): wave (ty) owns one 16-wide N tile.
// ---------------------------------------------------------------------------
__global__ void qkv_kernel(const float* __restrict__ x,
                           const float* __restrict__ W,
                           const float* __restrict__ bias,
                           float* __restrict__ dst) {
  const int lane = threadIdx.x;
  const int n0 = (blockIdx.x * blockDim.y + threadIdx.y) * 16;
  const int m0 = blockIdx.y * 16;
  v8f acc = {};
#pragma unroll 4
  for (int kk = 0; kk < HID; kk += 32) {
    v16h a = load_a(x + (size_t)m0 * HID, HID, kk, lane, 16);
    v16h b = load_b(W + (size_t)kk * HID + n0, HID, lane);
    acc = wmma_step(a, b, acc);
  }
  const int n  = n0 + (lane & 15);
  const int hh = n >> 6, dd = n & 63;
  const float bn = bias[n];
#pragma unroll
  for (int e = 0; e < 8; ++e) {
    const int m  = m0 + e + 8 * (lane >> 4);
    const int bb = m >> 9, l = m & (L - 1);
    dst[(((size_t)bb * H + hh) * L + l) * D + dd] = acc[e] + bn;
  }
}

// ---------------------------------------------------------------------------
// K2: content scores.  scores[bh,l,m] = Q[bh,l,:] . K[bh,m,:]
// grid (L/64, L/16, B*H), block (32,4).
// ---------------------------------------------------------------------------
__global__ void content_scores_kernel(const float* __restrict__ Q,
                                      const float* __restrict__ Kb,
                                      float* __restrict__ scores) {
  const int lane = threadIdx.x;
  const int bh = blockIdx.z;
  const int m0 = blockIdx.y * 16;
  const int n0 = (blockIdx.x * blockDim.y + threadIdx.y) * 16;
  const float* qp = Q  + (size_t)bh * L * D;
  const float* kp = Kb + (size_t)bh * L * D;
  v8f acc = {};
#pragma unroll
  for (int kk = 0; kk < D; kk += 32) {
    v16h a = load_a(qp + (size_t)m0 * D, D, kk, lane, 16);
    v16h b = load_bt(kp + (size_t)n0 * D + kk, D, lane);   // B = K^T
    acc = wmma_step(a, b, acc);
  }
  float* sp = scores + ((size_t)bh * L + m0) * L + n0;
  const int n = lane & 15;
#pragma unroll
  for (int e = 0; e < 8; ++e) {
    const int m = e + 8 * (lane >> 4);
    sp[(size_t)m * L + n] = acc[e];
  }
}

// ---------------------------------------------------------------------------
// K3: relation-K scores (head-shared GEMM per (b,l)), TDM-staged.
// scores[b,h,l,r] += Q[b,h,l,:] . rel_k[b,l,r,:]    A:[12(pad16) x 64]
// Each wave's B tile rel_k[b,l,n0:n0+16,:] is a contiguous 4KB slab
// (row stride == row length), DMA'd to a private LDS slice by the
// Tensor Data Mover, then consumed after s_wait_tensorcnt 0.
// grid (L/64, B*L), block (32,4).
// ---------------------------------------------------------------------------
__global__ void rel_scores_kernel(const float* __restrict__ Q,
                                  const float* __restrict__ relk,  // [B,L,L,D]
                                  float* __restrict__ scores) {
  __shared__ float btile[4 * 16 * D];   // 16KB: 4KB slice per wave
  const int lane = threadIdx.x;
  const int ty   = threadIdx.y;
  const int bl = blockIdx.y;
  const int b = bl >> 9, l = bl & (L - 1);
  const int n0 = (blockIdx.x * blockDim.y + ty) * 16;
  const float* qp = Q + ((size_t)b * H * L + l) * D;        // head stride L*D

  // ---- Build Tensor DMA Descriptor (D#) for a 1D copy of 1024 f32 ----
  float* myslice = &btile[ty * 16 * D];
  const unsigned long long ga =
      (unsigned long long)(const void*)(relk + ((size_t)bl * L + n0) * D);
  const unsigned int ga_lo  = (unsigned int)ga;
  const unsigned int ga_hi  = (unsigned int)(ga >> 32) & 0x01FFFFFFu;
  const unsigned int ldsoff = (unsigned int)(size_t)myslice;  // LDS byte offset
  // group0: count=1 | lds_addr | global_addr[31:0] | global_addr[56:32], type=2
  v4u g0 = { 0x1u, ldsoff, ga_lo, ga_hi | (2u << 30) };
  // group1: data_size=4B(code 2); tensor_dim0=1024; tensor_dim1=1;
  //         tile_dim0=1024; tile_dim1=1; dim0_stride=1024; dim1_stride=1024.
  v8i g1 = { 0x00020000,
             (int)(1024u << 16),          // tensor_dim0[15:0] in [31:16]
             (int)(1u << 16),             // tensor_dim1[15:0] in [31:16]
             (int)(1024u << 16),          // tile_dim0 in [31:16]
             1,                           // tile_dim1 = 1
             1024,                        // tensor_dim0_stride[31:0]
             (int)(1024u << 16),          // tensor_dim1_stride[15:0] in [31:16]
             0 };
  v4i g2 = { 0, 0, 0, 0 };
  v4i g3 = { 0, 0, 0, 0 };
#if defined(__clang_major__) && (__clang_major__ >= 23)
  v8i g4 = { 0, 0, 0, 0, 0, 0, 0, 0 };
  __builtin_amdgcn_tensor_load_to_lds(g0, g1, g2, g3, g4, 0);
#else
  __builtin_amdgcn_tensor_load_to_lds(g0, g1, g2, g3, 0);
#endif
  __builtin_amdgcn_s_wait_tensorcnt(0);   // wave-private slice: no barrier needed

  v8f acc = {};
#pragma unroll
  for (int kk = 0; kk < D; kk += 32) {
    v16h a  = load_a(qp, (long)L * D, kk, lane, H);         // rows >= 12 zero
    v16h b2 = load_bt(myslice + kk, D, lane);               // B = rel_k^T (LDS)
    acc = wmma_step(a, b2, acc);
  }
  const int n = lane & 15;
#pragma unroll
  for (int e = 0; e < 8; ++e) {
    const int h = e + 8 * (lane >> 4);
    if (h < H) {
      const size_t idx = (((size_t)b * H + h) * L + l) * L + n0 + n;
      scores[idx] += acc[e];
    }
  }
}

// ---------------------------------------------------------------------------
// K4: scaled softmax over last dim (512).  One block (256 threads) per row.
// ---------------------------------------------------------------------------
__global__ void softmax_kernel(float* __restrict__ scores) {
  const int t = threadIdx.x;
  float* p = scores + (size_t)blockIdx.x * L;
  const float scale = 0.125f;  // 1/sqrt(64)
  float a0 = p[t] * scale;
  float a1 = p[t + 256] * scale;
  __shared__ float red[256];
  red[t] = fmaxf(a0, a1);
  __syncthreads();
  for (int s = 128; s > 0; s >>= 1) {
    if (t < s) red[t] = fmaxf(red[t], red[t + s]);
    __syncthreads();
  }
  const float rowmax = red[0];
  __syncthreads();
  const float e0 = expf(a0 - rowmax);
  const float e1 = expf(a1 - rowmax);
  red[t] = e0 + e1;
  __syncthreads();
  for (int s = 128; s > 0; s >>= 1) {
    if (t < s) red[t] += red[t + s];
    __syncthreads();
  }
  const float inv = 1.0f / red[0];
  p[t]       = e0 * inv;
  p[t + 256] = e1 * inv;
}

// ---------------------------------------------------------------------------
// K5: content context.  out[b,l,h*64+d] = sum_m P[bh,l,m] * V[bh,m,d]
// Writes output (layout [B,L,HID]); K6 accumulates on top.
// grid (1, L/16, B*H), block (32,4): ty owns 16 of the 64 D columns.
// ---------------------------------------------------------------------------
__global__ void ctx_content_kernel(const float* __restrict__ P,
                                   const float* __restrict__ V,
                                   float* __restrict__ out) {
  const int lane = threadIdx.x;
  const int bh = blockIdx.z;
  const int m0 = blockIdx.y * 16;
  const int n0 = threadIdx.y * 16;
  const float* pp = P + ((size_t)bh * L + m0) * L;
  const float* vp = V + (size_t)bh * L * D + n0;
  v8f acc = {};
#pragma unroll 4
  for (int kk = 0; kk < L; kk += 32) {
    v16h a = load_a(pp, L, kk, lane, 16);
    v16h b = load_b(vp + (size_t)kk * D, D, lane);
    acc = wmma_step(a, b, acc);
  }
  const int bb = bh / H, h = bh % H;
  const int n = n0 + (lane & 15);
#pragma unroll
  for (int e = 0; e < 8; ++e) {
    const int l = m0 + e + 8 * (lane >> 4);
    out[((size_t)bb * L + l) * HID + h * D + n] = acc[e];
  }
}

// ---------------------------------------------------------------------------
// K6: relation-V context (head-shared GEMM per (b,l)).
// out[b,l,h*64+d] += sum_r P[b,h,l,r] * rel_v[b,r,l,d]
// rel_v rows strided by L*D (256B contiguous per r) — streamed once, prefetched.
// grid (1, B*L), block (32,4).
// ---------------------------------------------------------------------------
__global__ void ctx_rel_kernel(const float* __restrict__ P,
                               const float* __restrict__ relv,  // [B,L,L,D]
                               float* __restrict__ out) {
  const int lane = threadIdx.x;
  const int bl = blockIdx.y;
  const int b = bl >> 9, l = bl & (L - 1);
  const int n0 = threadIdx.y * 16;
  const float* pp = P + (size_t)b * H * L * L + (size_t)l * L;  // head stride L*L
  const float* vp = relv + ((size_t)b * L * L + l) * D + n0;    // r stride L*D
  v8f acc = {};
#pragma unroll 4
  for (int kk = 0; kk < L; kk += 32) {
    if (kk + 32 < L)
      __builtin_prefetch(vp + (size_t)(kk + 32) * (L * D), 0, 1);
    v16h a = load_a(pp, (long)L * L, kk, lane, H);              // rows >= 12 zero
    v16h bb2 = load_b(vp + (size_t)kk * (L * D), (long)L * D, lane);
    acc = wmma_step(a, bb2, acc);
  }
  const int n = n0 + (lane & 15);
#pragma unroll
  for (int e = 0; e < 8; ++e) {
    const int h = e + 8 * (lane >> 4);
    if (h < H) {
      const size_t idx = ((size_t)b * L + l) * HID + h * D + n;
      out[idx] += acc[e];
    }
  }
}

// ---------------------------------------------------------------------------
// Launcher.  Inputs (setup_inputs order):
// 0 hidden_states [B,L,HID], 1 rel_k [B,L,L,D], 2 rel_v [B,L,L,D],
// 3 Wq, 4 bq, 5 Wk, 6 bk, 7 Wv, 8 bv.   Output: [B,L,HID] fp32.
// Workspace: Q,K,V ([B,H,L,D] each) + scores [B,H,L,L]  = ~66 MB.
// ---------------------------------------------------------------------------
extern "C" void kernel_launch(void* const* d_in, const int* in_sizes, int n_in,
                              void* d_out, int out_size, void* d_ws, size_t ws_size,
                              hipStream_t stream) {
  (void)in_sizes; (void)n_in; (void)out_size; (void)ws_size;
  const float* x    = (const float*)d_in[0];
  const float* relk = (const float*)d_in[1];
  const float* relv = (const float*)d_in[2];
  const float* Wq = (const float*)d_in[3];
  const float* bq = (const float*)d_in[4];
  const float* Wk = (const float*)d_in[5];
  const float* bk = (const float*)d_in[6];
  const float* Wv = (const float*)d_in[7];
  const float* bv = (const float*)d_in[8];
  float* out = (float*)d_out;
  float* ws  = (float*)d_ws;

  const size_t QS = (size_t)BATCH * H * L * D;   // 1,572,864 floats
  float* Qb = ws;
  float* Kb = ws + QS;
  float* Vb = ws + 2 * QS;
  float* S  = ws + 3 * QS;                       // [B,H,L,L]

  const dim3 blk(32, 4);  // 4 waves of 32 (wave32)

  const dim3 gProj(HID / 64, (BATCH * L) / 16);
  qkv_kernel<<<gProj, blk, 0, stream>>>(x, Wq, bq, Qb);
  qkv_kernel<<<gProj, blk, 0, stream>>>(x, Wk, bk, Kb);
  qkv_kernel<<<gProj, blk, 0, stream>>>(x, Wv, bv, Vb);

  content_scores_kernel<<<dim3(L / 64, L / 16, BATCH * H), blk, 0, stream>>>(Qb, Kb, S);
  rel_scores_kernel<<<dim3(L / 64, BATCH * L), blk, 0, stream>>>(Qb, relk, S);
  softmax_kernel<<<dim3(BATCH * H * L), dim3(256), 0, stream>>>(S);
  ctx_content_kernel<<<dim3(1, L / 16, BATCH * H), blk, 0, stream>>>(S, Vb, out);
  ctx_rel_kernel<<<dim3(1, BATCH * L), blk, 0, stream>>>(S, relv, out);
}
